// ProtoPNet_6957847020234
// MI455X (gfx1250) — matrix operand order repaired and checked by reading, compiled, not run on hardware
//
#include <hip/hip_runtime.h>
#include <hip/hip_bf16.h>
#include <math.h>

typedef __attribute__((ext_vector_type(16))) _Float16 v16h;
typedef __attribute__((ext_vector_type(8)))  float    v8f;

#define BATCH   128
#define CIN     512
#define HW      196
#define NPAD    208      // 13 * 16
#define NTILES  13
#define DCH     128
#define NPROTO  2000
#define NCLS    200
#define MTILES_P 125     // 2000 / 16 exactly

// Fragment-layout sizes (halves)
#define FRAG_H       (32 * 16)                    // one 16x16x32 fragment: 32 lanes x 16 halves
#define BCHUNK_H     (NTILES * FRAG_H)            // one 32-deep k-chunk of B tiles (13 tiles)
#define BFULL_H      (4 * BCHUNK_H)               // K=128 B-matrix in frag layout (26624 halves)

// ---------------------------------------------------------------------------
// WMMA f16 16x16x32 layouts (wave32):
// A: lane l -> row = l&15 ; K = kc + (l>>4)*16 + {0..15}
// B: lane l -> col = l&15 ; K = kc + (l>>4)*16 + {0..15}
// C: vgpr r, lane l -> M = r + 8*(l>>4), N = l&15
// Fragment storage: [tile][lane][16 halves] => one contiguous 32B load per lane.
// ---------------------------------------------------------------------------

// ---------------------------------------------------------------------------
// Prep: pack f32 [M][K] row-major into f16 A-fragment layout [mt][kc][lane][j]
// ---------------------------------------------------------------------------
__global__ __launch_bounds__(256) void pp_pack_a(const float* __restrict__ src,
                                                 _Float16* __restrict__ dst,
                                                 int M, int K, int total) {
  const int KC = K >> 5;
  for (int idx = blockIdx.x * 256 + threadIdx.x; idx < total; idx += gridDim.x * 256) {
    const int j      = idx & 15;
    const int l      = (idx >> 4) & 31;
    const int rest   = idx >> 9;
    const int kc     = rest % KC;
    const int mt     = rest / KC;
    const int row    = mt * 16 + (l & 15);
    const int k      = kc * 32 + (l >> 4) * 16 + j; // contiguous: k = 16*half + j
    dst[idx] = (_Float16)src[(size_t)row * K + k];
  }
}

// Prep: p2[p] = ||prototype_p||^2
__global__ __launch_bounds__(256) void pp_p2(const float* __restrict__ protos,
                                             float* __restrict__ p2) {
  const int p = blockIdx.x * 256 + threadIdx.x;
  if (p >= NPROTO) return;
  const float* pr = protos + (size_t)p * DCH;
  float s = 0.0f;
#pragma unroll 8
  for (int k = 0; k < DCH; ++k) s = fmaf(pr[k], pr[k], s);
  p2[p] = s;
}

// ---------------------------------------------------------------------------
// Kernel 1: H1 = relu(W1 @ X_b + b1) -> stored f16 in B-fragment layout per b
// ---------------------------------------------------------------------------
__global__ __launch_bounds__(256) void pp_conv1(const float* __restrict__ x,
                                                const _Float16* __restrict__ w1f,
                                                const float* __restrict__ b1,
                                                _Float16* __restrict__ H1f) {
  __shared__ v16h Xs[NTILES * 32];          // one 32-deep k-chunk, frag layout (13.3 KB)
  const int b    = blockIdx.x;
  const int tid  = threadIdx.x;
  const int wave = tid >> 5;
  const int lane = tid & 31;
  const int lanelo = lane & 15, lanehi = lane >> 4;

  v8f acc[NTILES] = {};
  const float* xb = x + (size_t)b * CIN * HW;
  const v16h* af  = (const v16h*)w1f;
  _Float16* xsh   = (_Float16*)Xs;

  for (int kc = 0; kc < CIN / 32; ++kc) {
    // stage + permute X chunk into B-fragment layout
    for (int idx = tid; idx < 32 * NPAD; idx += 256) {
      const int kk = idx / NPAD, n = idx - kk * NPAD;
      const float v = (n < HW) ? xb[(size_t)(kc * 32 + kk) * HW + n] : 0.0f;
      const int l = ((kk >> 4) << 4) + (n & 15);          // (khalf*16 + lanelo)
      xsh[(((n >> 4) * 32) + l) * 16 + (kk & 15)] = (_Float16)v;
    }
    __syncthreads();

    const v16h afrag = af[(size_t)(wave * (CIN / 32) + kc) * 32 + lane];
#pragma unroll
    for (int t = 0; t < NTILES; ++t) {
      const v16h bfrag = Xs[t * 32 + lane];
      acc[t] = __builtin_amdgcn_wmma_f32_16x16x32_f16(
          false, afrag, false, bfrag, (short)0, acc[t], false, false);
    }
    __syncthreads();
  }

  // epilogue: bias + relu, scatter into B-fragment layout for conv2
  _Float16* out = H1f + (size_t)b * BFULL_H;
  const int kcO = wave >> 1, lO = ((wave & 1) << 4) + lanelo;
#pragma unroll
  for (int r = 0; r < 8; ++r) {
    const int m = wave * 16 + r + 8 * lanehi;
    const float bias = b1[m];
    const int j = r + 8 * lanehi;
#pragma unroll
    for (int t = 0; t < NTILES; ++t) {
      float v = acc[t][r] + bias;
      v = v > 0.0f ? v : 0.0f;
      out[(((size_t)kcO * NTILES + t) * 32 + lO) * 16 + j] = (_Float16)v;
    }
  }
}

// ---------------------------------------------------------------------------
// Kernel 2: F = sigmoid(W2 @ H1_b + b2) -> f16 frag layout; f2[n] (pad = 1e30)
// ---------------------------------------------------------------------------
__global__ __launch_bounds__(256) void pp_conv2(const _Float16* __restrict__ H1f,
                                                const _Float16* __restrict__ w2f,
                                                const float* __restrict__ b2,
                                                _Float16* __restrict__ Ff,
                                                float* __restrict__ f2ws) {
  __shared__ v16h Hs[4 * NTILES * 32];      // full K=128 B-matrix, frag layout (53 KB)
  __shared__ float f2s[NPAD];
  const int b    = blockIdx.x;
  const int tid  = threadIdx.x;
  const int wave = tid >> 5;
  const int lane = tid & 31;
  const int lanelo = lane & 15, lanehi = lane >> 4;

  { // contiguous vector copy: frag layout is linear in ws
    const uint4* src = (const uint4*)(H1f + (size_t)b * BFULL_H);
    uint4* dst = (uint4*)Hs;
    for (int i = tid; i < BFULL_H * 2 / 16; i += 256) dst[i] = src[i];
  }
  for (int i = tid; i < NPAD; i += 256) f2s[i] = 0.0f;
  __syncthreads();

  const v16h* af = (const v16h*)w2f;
  v8f acc[NTILES] = {};
#pragma unroll
  for (int ks = 0; ks < 4; ++ks) {
    const v16h afrag = af[(size_t)(wave * 4 + ks) * 32 + lane];
#pragma unroll
    for (int t = 0; t < NTILES; ++t) {
      const v16h bfrag = Hs[(ks * NTILES + t) * 32 + lane];
      acc[t] = __builtin_amdgcn_wmma_f32_16x16x32_f16(
          false, afrag, false, bfrag, (short)0, acc[t], false, false);
    }
  }

  _Float16* out = Ff + (size_t)b * BFULL_H;
  const int kcO = wave >> 1, lO = ((wave & 1) << 4) + lanelo;
#pragma unroll
  for (int t = 0; t < NTILES; ++t) {
    float ssq = 0.0f;
#pragma unroll
    for (int r = 0; r < 8; ++r) {
      const int m = wave * 16 + r + 8 * lanehi;
      const float z = acc[t][r] + b2[m];
      const float s = 1.0f / (1.0f + __expf(-z));
      out[(((size_t)kcO * NTILES + t) * 32 + lO) * 16 + (r + 8 * lanehi)] = (_Float16)s;
      ssq = fmaf(s, s, ssq);
    }
    atomicAdd(&f2s[t * 16 + lanelo], ssq);   // ds_add_f32
  }
  __syncthreads();
  for (int i = tid; i < NPAD; i += 256)
    f2ws[(size_t)b * NPAD + i] = (i < HW) ? f2s[i] : 1.0e30f;
}

// ---------------------------------------------------------------------------
// Kernel 3: min_d[b,p] = min_n ( f2[n] + p2[p] - 2 * <proto_p, F_b[:,n]> )
// grid = (16, 128); wave handles prototype M-tile = blockIdx.x*8 + wave
// ---------------------------------------------------------------------------
__global__ __launch_bounds__(256) void pp_proto(const _Float16* __restrict__ Ff,
                                                const _Float16* __restrict__ protof,
                                                const float* __restrict__ p2g,
                                                const float* __restrict__ f2ws,
                                                float* __restrict__ min_d) {
  __shared__ v16h Fs[4 * NTILES * 32];      // 53 KB
  __shared__ float p2s[128];
  const int mblock = blockIdx.x;
  const int b      = blockIdx.y;
  const int tid  = threadIdx.x;
  const int wave = tid >> 5;
  const int lane = tid & 31;
  const int lanelo = lane & 15, lanehi = lane >> 4;
  const int mtile = mblock * 8 + wave;

  {
    const uint4* src = (const uint4*)(Ff + (size_t)b * BFULL_H);
    uint4* dst = (uint4*)Fs;
    for (int i = tid; i < BFULL_H * 2 / 16; i += 256) dst[i] = src[i];
  }
  if (tid < 128) {
    const int p = mblock * 128 + tid;
    p2s[tid] = (p < NPROTO) ? p2g[p] : 0.0f;
  }
  __syncthreads();
  if (mtile >= MTILES_P) return;

  const v16h* af = (const v16h*)protof;
  v8f acc[NTILES] = {};
#pragma unroll
  for (int ks = 0; ks < 4; ++ks) {
    if (ks < 3)
      __builtin_prefetch(&af[(size_t)(mtile * 4 + ks + 1) * 32 + lane], 0, 0);
    const v16h afrag = af[(size_t)(mtile * 4 + ks) * 32 + lane];
#pragma unroll
    for (int t = 0; t < NTILES; ++t) {
      const v16h bfrag = Fs[(ks * NTILES + t) * 32 + lane];
      acc[t] = __builtin_amdgcn_wmma_f32_16x16x32_f16(
          false, afrag, false, bfrag, (short)0, acc[t], false, false);
    }
  }

  float f2loc[NTILES];
#pragma unroll
  for (int t = 0; t < NTILES; ++t)
    f2loc[t] = f2ws[(size_t)b * NPAD + t * 16 + lanelo];

#pragma unroll
  for (int r = 0; r < 8; ++r) {
    const int m = r + 8 * lanehi;
    const float p2v = p2s[wave * 16 + m];
    float mn = 1.0e38f;
#pragma unroll
    for (int t = 0; t < NTILES; ++t)
      mn = fminf(mn, f2loc[t] + p2v - 2.0f * acc[t][r]);
#pragma unroll
    for (int off = 8; off > 0; off >>= 1)
      mn = fminf(mn, __shfl_xor(mn, off, 32));
    if (lanelo == 0)
      min_d[(size_t)b * NPROTO + mtile * 16 + m] = mn;
  }
}

// ---------------------------------------------------------------------------
// Kernel 4: logits[b,c] = -sum_p min_d[b,p] * last_w[c,p]
// ---------------------------------------------------------------------------
__global__ __launch_bounds__(256) void pp_logits(const float* __restrict__ min_d,
                                                 const float* __restrict__ last_w,
                                                 float* __restrict__ logits) {
  const int idx = blockIdx.x * 256 + threadIdx.x;
  if (idx >= BATCH * NCLS) return;
  const int b = idx / NCLS, c = idx - b * NCLS;
  const float* md = min_d + (size_t)b * NPROTO;
  const float* lw = last_w + (size_t)c * NPROTO;
  float s = 0.0f;
#pragma unroll 4
  for (int p = 0; p < NPROTO; ++p) s = fmaf(md[p], lw[p], s);
  logits[idx] = -s;
}

// ---------------------------------------------------------------------------
extern "C" void kernel_launch(void* const* d_in, const int* in_sizes, int n_in,
                              void* d_out, int out_size, void* d_ws, size_t ws_size,
                              hipStream_t stream) {
  const float* x      = (const float*)d_in[0];
  const float* w1     = (const float*)d_in[1];
  const float* b1     = (const float*)d_in[2];
  const float* w2     = (const float*)d_in[3];
  const float* b2     = (const float*)d_in[4];
  const float* protos = (const float*)d_in[5];
  const float* last_w = (const float*)d_in[6];

  float* logits = (float*)d_out;                       // [128*200]
  float* min_d  = (float*)d_out + BATCH * NCLS;        // [128*2000]

  // Workspace layout (all offsets 32B aligned)
  char* ws = (char*)d_ws;
  const size_t fBytes = (size_t)BATCH * BFULL_H * sizeof(_Float16);  // 6.8 MB
  _Float16* Ff     = (_Float16*)(ws);
  _Float16* H1f    = (_Float16*)(ws + fBytes);
  _Float16* w1f    = (_Float16*)(ws + 2 * fBytes);                          // 128*512  f16
  _Float16* w2f    = (_Float16*)(ws + 2 * fBytes + 131072);                 // 128*128  f16
  _Float16* protof = (_Float16*)(ws + 2 * fBytes + 131072 + 32768);         // 2000*128 f16
  float*    p2     = (float*)   (ws + 2 * fBytes + 131072 + 32768 + 512000);
  float*    f2     = (float*)   (ws + 2 * fBytes + 131072 + 32768 + 512000 + 8000);

  // --- prep: pack A matrices to f16 fragment layout; prototype norms ---
  pp_pack_a<<<64, 256, 0, stream>>>(w1, w1f, DCH, CIN, DCH * CIN);
  pp_pack_a<<<64, 256, 0, stream>>>(w2, w2f, DCH, DCH, DCH * DCH);
  pp_pack_a<<<256, 256, 0, stream>>>(protos, protof, NPROTO, DCH, NPROTO * DCH);
  pp_p2<<<(NPROTO + 255) / 256, 256, 0, stream>>>(protos, p2);

  // --- main pipeline ---
  pp_conv1 <<<BATCH, 256, 0, stream>>>(x, w1f, b1, H1f);
  pp_conv2 <<<BATCH, 256, 0, stream>>>(H1f, w2f, b2, Ff, f2);
  pp_proto <<<dim3(16, BATCH), 256, 0, stream>>>(Ff, protof, p2, f2, min_d);
  pp_logits<<<(BATCH * NCLS + 255) / 256, 256, 0, stream>>>(min_d, last_w, logits);
}
